// AcousticFeedbackSim_61735859912773
// MI455X (gfx1250) — matrix operand order repaired and checked by reading, not compiled, not run on hardware
//
#include <hip/hip_runtime.h>

// ---------------------------------------------------------------------------
// AcousticFeedbackSim == causal 32768-tap FIR (16 batch rows) as implicit GEMM:
//   D[m=time, n=batch] = sum_j A[m,j] * B[j,n]
//   A = Toeplitz slice of h (precomputed table, WMMA A lane layout, L2-hot)
//   B = contiguous chunk of zero-padded input, TDM-staged into LDS and shared
//       by all 8 waves of the block (wave time-offset folded into A index)
// bf16 hi/lo split -> 3 x v_wmma_f32_16x16x32_bf16 per K=32 chunk (~fp32 acc).
// ---------------------------------------------------------------------------

typedef __attribute__((ext_vector_type(16))) __bf16          v16bf;
typedef __attribute__((ext_vector_type(16))) unsigned short  v16u;
typedef __attribute__((ext_vector_type(8)))  unsigned short  v8u;
typedef __attribute__((ext_vector_type(8)))  float           v8f;
typedef __attribute__((ext_vector_type(4)))  unsigned int    v4u;
typedef __attribute__((ext_vector_type(8)))  int             v8i;
typedef __attribute__((ext_vector_type(4)))  int             v4i;

#define NBATCH 16                 // batches == WMMA N dimension
#define TM     4                  // 16-sample tiles per wave
#define WAVES  8                  // waves per block (wave32)
#define GRP    16                 // chunks (K=32 each) staged per LDS group
#define SPAN   (16 * TM * WAVES)  // 512 output samples per block
#define QOFF   31                 // Toeplitz table offset: q = 2jc-4wv-r+QOFF

#if defined(__has_builtin)
#if __has_builtin(__builtin_amdgcn_tensor_load_to_lds)
#define HAVE_TDM 1
#else
#define HAVE_TDM 0
#endif
#else
#define HAVE_TDM 0
#endif

__device__ __forceinline__ unsigned short f2bf_rne(float f) {
  unsigned int u = __float_as_uint(f);
  unsigned int r = u + 0x7FFFu + ((u >> 16) & 1u);   // round-to-nearest-even
  return (unsigned short)(r >> 16);
}
__device__ __forceinline__ float bf2f(unsigned short h) {
  return __uint_as_float(((unsigned int)h) << 16);
}

// ---- pack input: f32 (B,T) -> zero-padded bf16 hi/lo (B, Tp), front pad = L
__global__ void afs_pack_x(const float* __restrict__ inp,
                           unsigned short* __restrict__ xh,
                           unsigned short* __restrict__ xl,
                           int T, int Tp, int PAD) {
  long idx = (long)blockIdx.x * blockDim.x + threadIdx.x;
  long total = (long)NBATCH * Tp;
  if (idx >= total) return;
  int  b = (int)(idx / Tp);
  long t = idx - (long)b * Tp;
  long s = t - PAD;
  float v = 0.0f;
  if (s >= 0 && s < T) v = inp[(long)b * T + s];
  unsigned short hi = f2bf_rne(v);
  xh[idx] = hi;
  xl[idx] = f2bf_rne(v - bf2f(hi));
}

// ---- Toeplitz A-operand table in 16x32 bf16 A lane layout, contiguous k = 16*(lane/16) + i
// G[q][lane][i]: element i of lane L -> (m = L&15, jj = k(L,i)),
// value h[L_taps + m - jj - 16*(q - QOFF)], zero outside [0, L_taps).
__global__ void afs_pack_h(const float* __restrict__ rir,
                           unsigned short* __restrict__ gh,
                           unsigned short* __restrict__ gl,
                           int L, int NQ) {
  int idx = blockIdx.x * blockDim.x + threadIdx.x;   // [0, NQ*32*16)
  if (idx >= NQ * 512) return;
  int i    = idx & 15;
  int lane = (idx >> 4) & 31;
  int q    = (idx >> 9) - QOFF;
  int m    = lane & 15;
  int k    = (lane < 16) ? i                          // lanes 0-15 : K 0-15
                         : i + 16;                    // lanes 16-31: K 16-31
  int hidx = L + m - k - 16 * q;
  float v = (hidx >= 0 && hidx < L) ? rir[hidx] : 0.0f;
  unsigned short hi = f2bf_rne(v);
  gh[idx] = hi;
  gl[idx] = f2bf_rne(v - bf2f(hi));
}

#if HAVE_TDM
// Issue one TDM 2D tile load: rows x tile_w bf16 elems, row stride in elems.
// D# bitfields per CDNA5 ISA ch.8.3/8.4 (count=1, type=2 "image", data_size=2B).
__device__ __forceinline__ void tdm_load_2d(const void* gtile, unsigned lds_byte,
                                            unsigned tile_w, unsigned rows,
                                            unsigned long long row_stride_elems) {
  unsigned long long ga = (unsigned long long)(uintptr_t)gtile;
  v4u g0;
  g0[0] = 1u;                                            // count=1 (valid, user)
  g0[1] = lds_byte;                                      // lds_addr (bytes)
  g0[2] = (unsigned)ga;                                  // global_addr[31:0]
  g0[3] = (unsigned)((ga >> 32) & 0x01FFFFFFu) | (2u << 30); // addr[56:32]|type=2
  const unsigned td0 = 0x40000000u;                      // huge dim0: no OOB clip
  v8i g1;
  g1[0] = (int)(1u << 16);                               // mask=0, data_size=1 (2B)
  g1[1] = (int)((td0 & 0xFFFFu) << 16);                  // tensor_dim0 lo16
  g1[2] = (int)(((td0 >> 16) & 0xFFFFu) | (rows << 16)); // dim0 hi | tensor_dim1 lo
  g1[3] = (int)(tile_w << 16);                           // dim1 hi=0 | tile_dim0
  g1[4] = (int)rows;                                     // tile_dim1 | tile_dim2=0
  g1[5] = (int)(unsigned)(row_stride_elems & 0xFFFFFFFFu);      // dim0_stride lo32
  g1[6] = (int)(unsigned)((row_stride_elems >> 32) & 0xFFFFu);  // dim0_stride hi16
  g1[7] = 0;
  v4i gz = {0, 0, 0, 0};
#if __clang_major__ >= 23
  v8i gz8 = {0, 0, 0, 0, 0, 0, 0, 0};
  __builtin_amdgcn_tensor_load_to_lds(g0, g1, gz, gz, gz8, 0);
#else
  __builtin_amdgcn_tensor_load_to_lds(g0, g1, gz, gz, 0);
#endif
}
#endif

// ---- main FIR-as-GEMM kernel: block shares B stream via TDM->LDS
__global__ void __launch_bounds__(WAVES * 32)
afs_fir_wmma(const unsigned short* __restrict__ xh,
             const unsigned short* __restrict__ xl,
             const unsigned short* __restrict__ gh,
             const unsigned short* __restrict__ gl,
             float* __restrict__ out, int T, int Tp, int ngrp) {
  // [double buffer][hi/lo][batch row][GRP*32 samples]  = 64 KB
  __shared__ __align__(128) unsigned short smem[2][2][NBATCH][GRP * 32];

  const int lane = threadIdx.x & 31;
  const int wv   = threadIdx.x >> 5;
  const long t0b = (long)blockIdx.x * SPAN;   // block's first output sample
  if (t0b >= T) return;

  const int b    = lane & 15;                 // batch column (WMMA N)
  const int hi16 = lane >> 4;                 // lane half -> B K-range
  const unsigned short* grh = gh + lane * 16;
  const unsigned short* grl = gl + lane * 16;

  v8f acc[TM] = {};

#if HAVE_TDM
  const unsigned lds_base = (unsigned)(uintptr_t)&smem[0][0][0][0];
  if (wv == 0) {  // prologue: group 0 -> buffer 0
    tdm_load_2d(xh + t0b, lds_base + 0u,     GRP * 32, NBATCH, (unsigned long long)Tp);
    tdm_load_2d(xl + t0b, lds_base + 16384u, GRP * 32, NBATCH, (unsigned long long)Tp);
  }
#endif

  for (int g = 0; g < ngrp; ++g) {
    const int buf = g & 1;
#if HAVE_TDM
    if (wv == 0) {
      if (g + 1 < ngrp) {   // keep next group's 2 TDM ops in flight
        const long xoff = t0b + (long)(g + 1) * (GRP * 32);
        const unsigned dst = lds_base + (unsigned)((g + 1) & 1) * 32768u;
        tdm_load_2d(xh + xoff, dst,           GRP * 32, NBATCH, (unsigned long long)Tp);
        tdm_load_2d(xl + xoff, dst + 16384u,  GRP * 32, NBATCH, (unsigned long long)Tp);
        __builtin_amdgcn_s_wait_tensorcnt(2);   // group g's pair complete (in-order)
      } else {
        __builtin_amdgcn_s_wait_tensorcnt(0);
      }
    }
    __syncthreads();        // LDS data for group g visible to all waves
#else
    { // fallback: cooperative synchronous staging (2048 x 16B by 256 threads)
      const long xoff = t0b + (long)g * (GRP * 32);
      for (int it = 0; it < 8; ++it) {
        int linear = (int)threadIdx.x + 256 * it;   // v8u units
        int sel = linear >> 10;
        int rem = linear & 1023;
        int row = rem >> 6;
        int col = (rem & 63) * 8;
        const unsigned short* src = (sel ? xl : xh) + (size_t)row * Tp + xoff + col;
        *(v8u*)&smem[buf][sel][row][col] = *(const v8u*)src;
      }
      __syncthreads();
    }
#endif

    const unsigned short* bhp = &smem[buf][0][b][hi16 * 16];
    const unsigned short* blp = &smem[buf][1][b][hi16 * 16];
    for (int jcg = 0; jcg < GRP; ++jcg) {
      v16bf bh = __builtin_bit_cast(v16bf, *(const v16u*)(bhp + 32 * jcg)); // ds_load
      v16bf bl = __builtin_bit_cast(v16bf, *(const v16u*)(blp + 32 * jcg));
      const int qbase = 2 * (g * GRP + jcg) + QOFF - 4 * wv;
#pragma unroll
      for (int r = 0; r < TM; ++r) {
        const size_t go = (size_t)(qbase - r) * 512;   // half-chunk Toeplitz shift
        v16bf Ah = __builtin_bit_cast(v16bf, *(const v16u*)(grh + go));
        v16bf Al = __builtin_bit_cast(v16bf, *(const v16u*)(grl + go));
        acc[r] = __builtin_amdgcn_wmma_f32_16x16x32_bf16(false, Ah, false, bh,
                                                         (short)0, acc[r], false, false);
        acc[r] = __builtin_amdgcn_wmma_f32_16x16x32_bf16(false, Ah, false, bl,
                                                         (short)0, acc[r], false, false);
        acc[r] = __builtin_amdgcn_wmma_f32_16x16x32_bf16(false, Al, false, bh,
                                                         (short)0, acc[r], false, false);
      }
    }
    __syncthreads();        // all waves done with buf before it is re-filled
  }

  // C/D layout: lane<16 rows 0..7, lane>=16 rows 8..15; 8 consecutive samples
  const long t0w = t0b + (long)wv * (16 * TM);
#pragma unroll
  for (int r = 0; r < TM; ++r) {
    float* op = out + (size_t)b * T + t0w + r * 16 + hi16 * 8;
    *(v8f*)op = acc[r];
  }
}

extern "C" void kernel_launch(void* const* d_in, const int* in_sizes, int n_in,
                              void* d_out, int out_size, void* d_ws, size_t ws_size,
                              hipStream_t stream) {
  const float* inp = (const float*)d_in[0];   // (16, T) f32
  const float* rir = (const float*)d_in[1];   // (L,)   f32, L = K*N taps
  (void)n_in; (void)out_size; (void)ws_size;

  const int T   = in_sizes[0] / NBATCH;       // 524288
  const int L   = in_sizes[1];                // 32768
  const int PAD = L;
  const int Tp  = ((PAD + T + 32 * GRP + 64) + 15) & ~15;  // padded, 32B rows

  // chunk count over the block-shared window [0, L + SPAN), rounded to groups
  int ncht = (L + SPAN + 31) / 32;
  ncht = ((ncht + GRP - 1) / GRP) * GRP;                    // 1040
  const int ngrp = ncht / GRP;                              // 65
  const int NQ   = 2 * (ncht - 1) + QOFF + 2;               // Toeplitz entries

  // Workspace layout
  unsigned short* xh = (unsigned short*)d_ws;
  unsigned short* xl = xh + (size_t)NBATCH * Tp;
  unsigned short* gh = xl + (size_t)NBATCH * Tp;
  unsigned short* gl = gh + (size_t)NQ * 512;

  { // 1) pack zero-padded bf16 hi/lo input
    long total = (long)NBATCH * Tp;
    afs_pack_x<<<(int)((total + 255) / 256), 256, 0, stream>>>(inp, xh, xl, T, Tp, PAD);
  }
  { // 2) build Toeplitz filter fragment table
    int total = NQ * 512;
    afs_pack_h<<<(total + 255) / 256, 256, 0, stream>>>(rir, gh, gl, L, NQ);
  }
  { // 3) FIR as implicit GEMM via WMMA + TDM double-buffered LDS staging
    int blocks = (T + SPAN - 1) / SPAN;                     // 1024
    afs_fir_wmma<<<blocks, WAVES * 32, 0, stream>>>(xh, xl, gh, gl,
                                                    (float*)d_out, T, Tp, ngrp);
  }
}
